// CausalMultiHeadAttention_63857573757494
// MI455X (gfx1250) — compile-verified, matrix operand order checked by reading, not run on hardware
//
#include <hip/hip_runtime.h>

typedef __attribute__((ext_vector_type(16))) __bf16          v16bf;
typedef __attribute__((ext_vector_type(16))) unsigned short  v16u;
typedef __attribute__((ext_vector_type(8)))  float           v8f;
typedef int v4i __attribute__((vector_size(16)));

#define BH    64        // B*H
#define SQ    512       // sequence length
#define DIM   512
#define DK    64
#define KCAP  1024
#define MROWS 4096      // B*S

#define AS1 __attribute__((address_space(1)))
#define AS3 __attribute__((address_space(3)))

#if defined(__has_builtin)
#  if __has_builtin(__builtin_amdgcn_global_load_async_to_lds_b128)
#    define HAVE_ASYNC_LDS 1
#  endif
#endif

// float -> bf16 (round-to-nearest-even) as raw u16
__device__ __forceinline__ unsigned short f2bf(float f) {
  unsigned int u = __float_as_uint(f);
  u += 0x7FFFu + ((u >> 16) & 1u);
  return (unsigned short)(u >> 16);
}

// 16-byte global->LDS copy: async (ASYNCcnt, no VGPR transit) when available.
__device__ __forceinline__ void cp16(unsigned short* l, const unsigned short* g) {
#ifdef HAVE_ASYNC_LDS
  __builtin_amdgcn_global_load_async_to_lds_b128((AS1 v4i*)g, (AS3 v4i*)l, 0, 0);
#else
  *(int4*)l = *(const int4*)g;
#endif
}

template <int N>
__device__ __forceinline__ void wait_async() {
#ifdef HAVE_ASYNC_LDS
#  if __has_builtin(__builtin_amdgcn_s_wait_asynccnt)
  __builtin_amdgcn_s_wait_asynccnt(N);
#  else
  asm volatile("s_wait_asynccnt %0" :: "n"(N) : "memory");
#  endif
#endif
}

// K-index inside a 16x32 16-bit A/B tile for element j of the v16 register,
// given which half (lanes 0-15 vs 16-31) this lane is in.  (ISA 7.12.2)
__device__ __forceinline__ int kmap(int j, int hh) {
  int vg = j >> 1, pos = j & 1;
  int base = hh * 8;
  return (vg < 4) ? (base + vg * 2 + pos) : (16 + base + (vg - 4) * 2 + pos);
}

// ---------------------------------------------------------------------------
// One-time fp32 -> bf16 conversion pass.
__global__ __launch_bounds__(256) void convert_bf16_kernel(
    const float* __restrict__ src, unsigned short* __restrict__ dst, int n) {
  int i = blockIdx.x * 256 + threadIdx.x;
  if (i < n) dst[i] = f2bf(src[i]);
}

// KV cache shift: dst rows [0,512) <- prev rows [512,1024), per (b,h) block,
// written both as f32 (final output cache) and bf16 (attention operand).
__global__ __launch_bounds__(256) void shift_copy_kernel(
    const float* __restrict__ pk, const float* __restrict__ pv,
    float* __restrict__ ko, float* __restrict__ vo,
    unsigned short* __restrict__ kbf, unsigned short* __restrict__ vbf) {
  long i  = (long)blockIdx.x * 256 + threadIdx.x;  // 0 .. 2097151
  long bh = i >> 15;
  long j  = i & 32767;
  long src = bh * 65536 + 32768 + j;
  long dst = bh * 65536 + j;
  float kv = pk[src], vv = pv[src];
  ko[dst] = kv;        vo[dst] = vv;
  kbf[dst] = f2bf(kv); vbf[dst] = f2bf(vv);
}

__global__ void set_offset_kernel(int* p, int idx, int val) {
  if (threadIdx.x == 0 && blockIdx.x == 0) p[idx] = val;
}

// ---------------------------------------------------------------------------
// C = X(MxK) @ W^T(NxK) + bias; X, W already bf16. Double-buffered async LDS
// staging, 64x64 block tile, 8 waves, v_wmma_f32_16x16x32_bf16.
// MODE 0: bf16 (B,H,S,DK) q buffer    MODE 1: f32 + bf16 cache rows 512..1023
// MODE 2: f32 row-major
__device__ __forceinline__ void stage_xw(
    unsigned short* xd, unsigned short* wd,
    const unsigned short* xs, const unsigned short* ws,
    int k0, int tid, int K) {
  int row = tid >> 2, ch = (tid & 3) * 8;
  long s = (long)row * K + k0 + ch;
  int d = tid * 8;
  cp16(xd + d, xs + s);
  cp16(wd + d, ws + s);
}

template <int MODE>
__global__ __launch_bounds__(256) void gemm_bias_kernel(
    const unsigned short* __restrict__ X, const unsigned short* __restrict__ W,
    const float* __restrict__ bias, float* __restrict__ out,
    unsigned short* __restrict__ bfout, int M, int N, int K) {
  __shared__ alignas(16) unsigned short Xs[2][64 * 32];
  __shared__ alignas(16) unsigned short Wsb[2][64 * 32];

  const int tid  = threadIdx.x;
  const int lane = tid & 31;
  const int wv   = tid >> 5;        // 0..7
  const int mi   = wv & 3;          // 16-row chunk
  const int ni   = wv >> 2;         // 32-col chunk
  const int hh   = (lane >> 4) & 1;
  const int ln   = lane & 15;
  const int m0   = blockIdx.x * 64;
  const int n0   = blockIdx.y * 64;

  const unsigned short* xsrc = X + (long)m0 * K;
  const unsigned short* wsrc = W + (long)n0 * K;

  v8f acc0 = {};
  v8f acc1 = {};

  stage_xw(Xs[0], Wsb[0], xsrc, wsrc, 0, tid, K);

  const int nks = K >> 5;
  for (int ksi = 0; ksi < nks; ++ksi) {
    const int cur = ksi & 1;
    if (ksi + 1 < nks) {
      stage_xw(Xs[cur ^ 1], Wsb[cur ^ 1], xsrc, wsrc, (ksi + 1) * 32, tid, K);
      wait_async<2>();
    } else {
      wait_async<0>();
    }
    __syncthreads();
    const unsigned short* Xc = Xs[cur];
    const unsigned short* Wc = Wsb[cur];

    v16u au;
#pragma unroll
    for (int j = 0; j < 16; ++j)
      au[j] = Xc[(mi * 16 + ln) * 32 + kmap(j, hh)];
    v16bf a = __builtin_bit_cast(v16bf, au);

#pragma unroll
    for (int t = 0; t < 2; ++t) {
      v16u bu;
#pragma unroll
      for (int j = 0; j < 16; ++j)
        bu[j] = Wc[(ni * 32 + t * 16 + ln) * 32 + kmap(j, hh)];
      v16bf bm = __builtin_bit_cast(v16bf, bu);
      if (t == 0)
        acc0 = __builtin_amdgcn_wmma_f32_16x16x32_bf16(false, a, false, bm,
                                                       (short)0, acc0, false, false);
      else
        acc1 = __builtin_amdgcn_wmma_f32_16x16x32_bf16(false, a, false, bm,
                                                       (short)0, acc1, false, false);
    }
    __syncthreads();
  }

#pragma unroll
  for (int t = 0; t < 2; ++t) {
    v8f acc = t ? acc1 : acc0;
    int c = n0 + ni * 32 + t * 16 + ln;
    float bvl = bias[c];
#pragma unroll
    for (int r = 0; r < 8; ++r) {
      int row = m0 + mi * 16 + hh * 8 + r;
      float v = acc[r] + bvl;
      if constexpr (MODE == 2) {
        out[(long)row * N + c] = v;
      } else {
        int bb = row >> 9, s = row & 511;
        int hd = c >> 6, d = c & 63;
        if constexpr (MODE == 0) {
          bfout[((long)(bb * 8 + hd) * SQ + s) * DK + d] = f2bf(v);
        } else {
          long off = ((long)(bb * 8 + hd) * KCAP + 512 + s) * DK + d;
          out[off]   = v;
          bfout[off] = f2bf(v);
        }
      }
    }
  }
}

// ---------------------------------------------------------------------------
// Flash attention over bf16 Q/K/V. 1 block = (b, h, 64 q rows), 128 threads =
// 4 waves; double-buffered async K/V tiles (32 keys x 64 d, bf16).
__device__ __forceinline__ void stage_kv(
    unsigned short* kd, unsigned short* vd,
    const unsigned short* ks, const unsigned short* vs, int tid) {
#pragma unroll
  for (int r = 0; r < 2; ++r) {
    int e = (tid + r * 128) * 8;
    cp16(kd + e, ks + e);
    cp16(vd + e, vs + e);
  }
}

__global__ __launch_bounds__(128) void attn_kernel(
    const unsigned short* __restrict__ qbf, const unsigned short* __restrict__ kbf,
    const unsigned short* __restrict__ vbf, unsigned short* __restrict__ x2b) {
  __shared__ alignas(16) unsigned short Ks[2][32 * 64];
  __shared__ alignas(16) unsigned short Vs[2][32 * 64];
  __shared__ alignas(16) unsigned short Ps[4][16 * 32];

  const int bid = blockIdx.x;
  const int qt  = bid & 7;
  const int hd  = (bid >> 3) & 7;
  const int bb  = bid >> 6;
  const int s0  = qt * 64;

  const int tid  = threadIdx.x;
  const int lane = tid & 31;
  const int wv   = tid >> 5;      // 0..3
  const int hh   = (lane >> 4) & 1;
  const int ln   = lane & 15;
  const int srow = s0 + wv * 16;

  // Preload Q tile (16x64 bf16) into two 16x32 A-register tiles.
  v16bf aq[2];
  long qbase = ((long)(bb * 8 + hd) * SQ + (srow + ln)) * DK;
#pragma unroll
  for (int c = 0; c < 2; ++c) {
    v16u au;
#pragma unroll
    for (int j = 0; j < 16; ++j)
      au[j] = qbf[qbase + c * 32 + kmap(j, hh)];
    aq[c] = __builtin_bit_cast(v16bf, au);
  }

  float mrun[8], lrun[8];
  v8f acc[4];
  v8f zero = {};
#pragma unroll
  for (int r = 0; r < 8; ++r) { mrun[r] = -1e30f; lrun[r] = 0.0f; }
#pragma unroll
  for (int dt = 0; dt < 4; ++dt) acc[dt] = zero;

  const int nkb = (s0 + 576) >> 5;                 // key blocks of 32
  const unsigned short* ksrc = kbf + (long)(bb * 8 + hd) * KCAP * DK;
  const unsigned short* vsrc = vbf + (long)(bb * 8 + hd) * KCAP * DK;

  stage_kv(Ks[0], Vs[0], ksrc, vsrc, tid);

  for (int kb = 0; kb < nkb; ++kb) {
    const int cur  = kb & 1;
    const int key0 = kb * 32;
    if (kb + 1 < nkb) {
      long off = (long)(kb + 1) * 32 * DK;
      stage_kv(Ks[cur ^ 1], Vs[cur ^ 1], ksrc + off, vsrc + off, tid);
      wait_async<4>();
    } else {
      wait_async<0>();
    }
    __syncthreads();
    const unsigned short* Kc = Ks[cur];
    const unsigned short* Vc = Vs[cur];

    // Scores S = Q @ K^T  (16 x 32), two 16x16 column tiles.
    v8f sc[2];
#pragma unroll
    for (int t = 0; t < 2; ++t) {
      v8f s8 = {};
#pragma unroll
      for (int c = 0; c < 2; ++c) {
        v16u bu;
#pragma unroll
        for (int j = 0; j < 16; ++j)
          bu[j] = Kc[(t * 16 + ln) * 64 + c * 32 + kmap(j, hh)];
        v16bf bm = __builtin_bit_cast(v16bf, bu);
        s8 = __builtin_amdgcn_wmma_f32_16x16x32_bf16(false, aq[c], false, bm,
                                                     (short)0, s8, false, false);
      }
      sc[t] = s8;
    }

    // Scale (dk^-0.5 = 0.125) + causal mask (attend iff key <= s + 512).
#pragma unroll
    for (int t = 0; t < 2; ++t) {
      int key = key0 + t * 16 + ln;
#pragma unroll
      for (int r = 0; r < 8; ++r) {
        int sidx = srow + hh * 8 + r;
        float v = sc[t][r] * 0.125f;
        sc[t][r] = (key > sidx + 512) ? -1e30f : v;
      }
    }

    // Online softmax (row reductions across the 16 lanes holding that row).
    float alpha[8];
#pragma unroll
    for (int r = 0; r < 8; ++r) {
      float rm = fmaxf(sc[0][r], sc[1][r]);
#pragma unroll
      for (int m = 1; m < 16; m <<= 1) rm = fmaxf(rm, __shfl_xor(rm, m, 32));
      float mn = fmaxf(mrun[r], rm);
      alpha[r] = __expf(mrun[r] - mn);
      mrun[r] = mn;
      float p0 = __expf(sc[0][r] - mn);
      float p1 = __expf(sc[1][r] - mn);
      sc[0][r] = p0;
      sc[1][r] = p1;
      float rs = p0 + p1;
#pragma unroll
      for (int m = 1; m < 16; m <<= 1) rs += __shfl_xor(rs, m, 32);
      lrun[r] = lrun[r] * alpha[r] + rs;
    }
#pragma unroll
    for (int dt = 0; dt < 4; ++dt)
#pragma unroll
      for (int r = 0; r < 8; ++r) acc[dt][r] *= alpha[r];

    // P (C-layout) -> per-wave LDS -> A-layout bf16 registers.
#pragma unroll
    for (int t = 0; t < 2; ++t)
#pragma unroll
      for (int r = 0; r < 8; ++r)
        Ps[wv][(hh * 8 + r) * 32 + t * 16 + ln] = f2bf(sc[t][r]);
    __builtin_amdgcn_wave_barrier();
    asm volatile("s_wait_dscnt 0" ::: "memory");  // in-wave DS store->load

    v16u pu;
#pragma unroll
    for (int j = 0; j < 16; ++j) pu[j] = Ps[wv][ln * 32 + kmap(j, hh)];
    v16bf pa = __builtin_bit_cast(v16bf, pu);

    // O += P @ V  (16x32 @ 32x64), four 16x16 output tiles over d.
#pragma unroll
    for (int dt = 0; dt < 4; ++dt) {
      v16u bu;
#pragma unroll
      for (int j = 0; j < 16; ++j)
        bu[j] = Vc[kmap(j, hh) * 64 + dt * 16 + ln];
      v16bf bm = __builtin_bit_cast(v16bf, bu);
      acc[dt] = __builtin_amdgcn_wmma_f32_16x16x32_bf16(false, pa, false, bm,
                                                        (short)0, acc[dt], false, false);
    }
    __syncthreads();
  }

  // Normalize and scatter to bf16 (B, S, H*DK) intermediate.
#pragma unroll
  for (int dt = 0; dt < 4; ++dt) {
#pragma unroll
    for (int r = 0; r < 8; ++r) {
      int s = srow + hh * 8 + r;
      int d = hd * DK + dt * 16 + ln;
      x2b[((long)bb * SQ + s) * DIM + d] = f2bf(acc[dt][r] / lrun[r]);
    }
  }
}

// ---------------------------------------------------------------------------
extern "C" void kernel_launch(void* const* d_in, const int* in_sizes, int n_in,
                              void* d_out, int out_size, void* d_ws, size_t ws_size,
                              hipStream_t stream) {
  (void)in_sizes; (void)n_in; (void)ws_size; (void)out_size;

  const float* x  = (const float*)d_in[0];
  const float* pk = (const float*)d_in[1];
  const float* pv = (const float*)d_in[2];
  const float* Wq = (const float*)d_in[3];
  const float* bq = (const float*)d_in[4];
  const float* Wk = (const float*)d_in[5];
  const float* bk = (const float*)d_in[6];
  const float* Wv = (const float*)d_in[7];
  const float* bv = (const float*)d_in[8];
  const float* Wo = (const float*)d_in[9];
  const float* bo = (const float*)d_in[10];

  const long OUT_N = (long)MROWS * DIM;            // 2,097,152
  const long KV_N  = (long)BH * KCAP * DK;         // 4,194,304
  const long W_N   = (long)DIM * DIM;              // 262,144

  float* out  = (float*)d_out;
  float* kout = out + OUT_N;
  float* vout = kout + KV_N;

  unsigned short* xbf = (unsigned short*)d_ws;     // bf16 scratch layout
  unsigned short* wqb = xbf + OUT_N;
  unsigned short* wkb = wqb + W_N;
  unsigned short* wvb = wkb + W_N;
  unsigned short* wob = wvb + W_N;
  unsigned short* qbf = wob + W_N;                 // (B,H,S,DK)
  unsigned short* kbf = qbf + OUT_N;               // (B,H,KCAP,DK)
  unsigned short* vbf = kbf + KV_N;
  unsigned short* x2b = vbf + KV_N;                // (B,S,D)

  // 0) one-time bf16 conversions of x and the four weight matrices
  convert_bf16_kernel<<<(int)(OUT_N / 256), 256, 0, stream>>>(x, xbf, (int)OUT_N);
  convert_bf16_kernel<<<(int)(W_N / 256), 256, 0, stream>>>(Wq, wqb, (int)W_N);
  convert_bf16_kernel<<<(int)(W_N / 256), 256, 0, stream>>>(Wk, wkb, (int)W_N);
  convert_bf16_kernel<<<(int)(W_N / 256), 256, 0, stream>>>(Wv, wvb, (int)W_N);
  convert_bf16_kernel<<<(int)(W_N / 256), 256, 0, stream>>>(Wo, wob, (int)W_N);

  // 1) shift KV cache (rows 0..511), f32 + bf16
  shift_copy_kernel<<<8192, 256, 0, stream>>>(pk, pv, kout, vout, kbf, vbf);

  // 2) projections
  dim3 ggrid(MROWS / 64, DIM / 64);
  gemm_bias_kernel<0><<<ggrid, 256, 0, stream>>>(xbf, wqb, bq, nullptr, qbf,
                                                 MROWS, DIM, DIM);
  gemm_bias_kernel<1><<<ggrid, 256, 0, stream>>>(xbf, wkb, bk, kout, kbf,
                                                 MROWS, DIM, DIM);
  gemm_bias_kernel<1><<<ggrid, 256, 0, stream>>>(xbf, wvb, bv, vout, vbf,
                                                 MROWS, DIM, DIM);

  // 3) flash attention over the fully assembled bf16 caches
  attn_kernel<<<BH * (SQ / 64), 128, 0, stream>>>(qbf, kbf, vbf, x2b);

  // 4) output projection (f32 result)
  gemm_bias_kernel<2><<<ggrid, 256, 0, stream>>>(x2b, wob, bo, out, nullptr,
                                                 MROWS, DIM, DIM);

  // 5) new_offset = min(1024 + 512, 1024) = 1024, int32 in the tail slot
  set_offset_kernel<<<1, 1, 0, stream>>>((int*)d_out, (int)(OUT_N + 2 * KV_N), 1024);
}
